// AttentionBlock_12000138625524
// MI455X (gfx1250) — matrix operand order repaired and checked by reading, not compiled, not run on hardware
//
#include <hip/hip_runtime.h>
#include <hip/hip_bf16.h>

typedef __attribute__((ext_vector_type(16))) __bf16 v16bf;
typedef __attribute__((ext_vector_type(8)))  float  v8f;

#define NTOK 4096   // H*W
#define CCH  256    // C
#define NB   4      // B

// ---------------------------------------------------------------------------
// Kernel 1: 1x1-conv projections q,k,v  -> bf16, pre-swizzled into CDNA5 WMMA
// operand layouts so the attention kernel does pure contiguous 32B loads/lane.
//
//  qA[b][it=N/16][lane=32][e=16] : A-layout of Q^T tile (M=i rows, K=c8 chans)
//  kB[b][jt=N/16][lane=32][e=16] : B-layout of K tile   (K=c8 chans, N=j cols)
//  vB[b][jb=N/32][ct=C/16][lane=32][e=16] : B-layout of V^T (K=j, N=c cols)
// ---------------------------------------------------------------------------
__global__ void proj_swizzle_kernel(const float* __restrict__ x,
                                    const float* __restrict__ wq, const float* __restrict__ bq,
                                    const float* __restrict__ wk, const float* __restrict__ bk,
                                    const float* __restrict__ wv, const float* __restrict__ bv,
                                    __bf16* __restrict__ qA,
                                    __bf16* __restrict__ kB,
                                    __bf16* __restrict__ vB)
{
    const int blk = blockIdx.x;
    const int b   = blk / 320;          // 320 = 32(q) + 32(k) + 256(v) rows
    const int row = blk % 320;
    const int t   = threadIdx.x;        // 256 threads, each handles 16 columns

    const float* w; float bias; int mode, o;
    if (row < 32)      { mode = 0; o = row;      w = wq + o * CCH; bias = bq[o]; }
    else if (row < 64) { mode = 1; o = row - 32; w = wk + o * CCH; bias = bk[o]; }
    else               { mode = 2; o = row - 64; w = wv + o * CCH; bias = bv[o]; }

    float acc[16];
#pragma unroll
    for (int u = 0; u < 16; ++u) acc[u] = 0.f;

    const float* xb = x + (size_t)b * CCH * NTOK;
    for (int c = 0; c < CCH; ++c) {
        const float wc = w[c];                    // uniform -> scalar load
        const float* xr = xb + (size_t)c * NTOK + t;
#pragma unroll
        for (int u = 0; u < 16; ++u) acc[u] += wc * xr[u * 256];
    }

#pragma unroll
    for (int u = 0; u < 16; ++u) {
        const int n = u * 256 + t;
        const float val = acc[u] + bias;
        const __bf16 h = (__bf16)val;
        if (mode == 0) {
            const int it = n >> 4, m = n & 15;
            const int group = (o >> 3) & 1;
            const int p = (o < 16) ? ((o - 8 * group) >> 1)
                                   : (4 + ((o - 16 - 8 * group) >> 1));
            const int e = 2 * p + (o & 1);
            qA[(((size_t)b * 256 + it) * 32 + group * 16 + m) * 16 + e] = h;
        } else if (mode == 1) {
            const int jt = n >> 4, col = n & 15;
            const int group = (o >> 3) & 1, e = (o & 7) | ((o >> 4) << 3);
            kB[(((size_t)b * 256 + jt) * 32 + group * 16 + col) * 16 + e] = h;
        } else {
            const int jb = n >> 5, group = (n >> 3) & 1, e = (n & 7) | (((n >> 4) & 1) << 3);
            const int ct = o >> 4, colc = o & 15;
            vB[((((size_t)b * 128 + jb) * 16 + ct) * 32 + group * 16 + colc) * 16 + e] = h;
        }
    }
}

// ---------------------------------------------------------------------------
// Tensor Data Mover: DMA one 16KB V tile (16 c-tiles x 512 bf16) into LDS.
// D# built per CDNA5 ISA: group0 = control/lds/global addr,
// group1 = data_size/dims/strides, groups2-3 zero (<=2D tensor).
// This toolchain exposes the 6-arg builtin:
//   (uint32x4 g0, int32x8 g1, int32x4 g2, int32x4 g3, int32x8 extra, i32 cpol)
// ---------------------------------------------------------------------------
#if __has_builtin(__builtin_amdgcn_tensor_load_to_lds)
#define HAS_TDM 1
typedef __attribute__((ext_vector_type(4))) unsigned int v4u;
typedef __attribute__((ext_vector_type(8))) int v8i;
typedef __attribute__((ext_vector_type(4))) int v4i;

__device__ __forceinline__ void tdm_load_vtile(const __bf16* gsrc, unsigned lds_off)
{
    const unsigned long long ga = (unsigned long long)(uintptr_t)gsrc;
    v4u g0;
    g0[0] = 1u;                                    // count=1 (valid), user mode
    g0[1] = lds_off;                               // lds_addr (bytes)
    g0[2] = (unsigned)(ga & 0xffffffffu);          // global_addr[31:0]
    g0[3] = (unsigned)((ga >> 32) & 0x01ffffffu)   // global_addr[56:32]
          | 0x80000000u;                           // type=2 ("image")
    v8i g1;
    g1[0] = (int)(3u << 16);                       // data_size=8B; no flags
    g1[1] = (int)(2048u << 16);                    // tensor_dim0=2048 (lo16 @63:48)
    g1[2] = (int)(1u << 16);                       // tensor_dim0 hi=0 | tensor_dim1=1
    g1[3] = (int)(2048u << 16);                    // tensor_dim1 hi=0 | tile_dim0=2048
    g1[4] = 1;                                     // tile_dim1=1, tile_dim2=0
    g1[5] = 2048;                                  // tensor_dim0_stride lo32
    g1[6] = (int)(2048u << 16);                    // stride0 hi=0 | stride1 lo16
    g1[7] = 0;
    const v4i gz4 = {0, 0, 0, 0};
    const v8i gz8 = {0, 0, 0, 0, 0, 0, 0, 0};
    __builtin_amdgcn_tensor_load_to_lds(g0, g1, gz4, gz4, gz8, 0);
}
#else
#define HAS_TDM 0
#endif

// ---------------------------------------------------------------------------
// Kernel 2: two-pass exact-softmax attention over N=4096 keys per 16-row
// query tile. 4 waves/block; wave owns a 16x256 fp32 accumulator.
// Pass 1: energy WMMAs + per-lane row-max only (no cross-lane ops in loop).
// Pass 2: TDM double-buffers V tiles into LDS (one DMA per block per j-block)
//         while waves run energy WMMAs, exp, LDS transpose, 16 PV WMMAs.
// ---------------------------------------------------------------------------
__global__ void __launch_bounds__(128, 1)
flash_attn_kernel(const __bf16* __restrict__ qA,
                  const __bf16* __restrict__ kB,
                  const __bf16* __restrict__ vB,
                  const float*  __restrict__ x,
                  const float*  __restrict__ gamma,
                  float* __restrict__ out)
{
    __shared__ __align__(32) __bf16 psh[4][512];   // per-wave 16x32 P tile (A-layout)
#if HAS_TDM
    __shared__ __align__(32) __bf16 vlds[2][8192]; // double-buffered 16KB V tiles
#endif

    const int lane  = threadIdx.x & 31;
    const int wave  = threadIdx.x >> 5;
    const int b     = blockIdx.x >> 6;             // 64 i-blocks per batch
    const int it    = (blockIdx.x & 63) * 4 + wave;
    const int group = lane >> 4;
    const int lm    = lane & 15;

    // Q^T tile, already in A-layout: one 32B load per lane
    const v16bf qa = *(const v16bf*)(qA + (((size_t)b * 256 + it) * 32 + lane) * 16);
    const v8f vzero = {};

    // ---- precompute jb-invariant LDS scatter addresses (D-layout -> A-layout)
    __bf16* myp = &psh[wave][0];
    __bf16* a0[8];
    __bf16* a1[8];
#pragma unroll
    for (int r = 0; r < 8; ++r) {
        const int rowi = r + 8 * group;
        {
            const int K = lm;                       // j col 0..15
            const int gr = (K >> 3) & 1;
            const int pp = (K - 8 * gr) >> 1;
            const int e  = 2 * pp + (K & 1);
            a0[r] = myp + (rowi + 16 * gr) * 16 + e;
        }
        {
            const int K = 16 + lm;                  // j col 16..31
            const int gr = (K >> 3) & 1;
            const int pp = 4 + ((K - 16 - 8 * gr) >> 1);
            const int e  = 2 * pp + (K & 1);
            a1[r] = myp + (rowi + 16 * gr) * 16 + e;
        }
    }

    const __bf16* kbase = kB + (((size_t)b * 256) * 32 + lane) * 16;

    // =======================  PASS 1: global row max  =======================
    float mloc[8];
#pragma unroll
    for (int r = 0; r < 8; ++r) mloc[r] = -__builtin_inff();

    {
        const __bf16* kptr = kbase;
        for (int jb = 0; jb < 128; ++jb) {
            const v16bf kb0 = *(const v16bf*)kptr;
            const v16bf kb1 = *(const v16bf*)(kptr + 512);
            kptr += 1024;
            v8f e0 = __builtin_amdgcn_wmma_f32_16x16x32_bf16(false, qa, false, kb0,
                                                             (short)0, vzero, false, false);
            v8f e1 = __builtin_amdgcn_wmma_f32_16x16x32_bf16(false, qa, false, kb1,
                                                             (short)0, vzero, false, false);
#pragma unroll
            for (int r = 0; r < 8; ++r)
                mloc[r] = fmaxf(mloc[r], fmaxf(e0[r], e1[r]));
        }
    }
    // batched butterfly reduction across the 16 column-lanes (bpermutes overlap)
#pragma unroll
    for (int s = 1; s < 16; s <<= 1) {
        float t[8];
#pragma unroll
        for (int r = 0; r < 8; ++r) t[r] = __shfl_xor(mloc[r], s, 32);
#pragma unroll
        for (int r = 0; r < 8; ++r) mloc[r] = fmaxf(mloc[r], t[r]);
    }

    // ==================  PASS 2: P = exp(E - m), O = P V  ===================
    v8f acc[16];
#pragma unroll
    for (int ct = 0; ct < 16; ++ct) acc[ct] = vzero;

    float lsum[8];
#pragma unroll
    for (int r = 0; r < 8; ++r) lsum[r] = 0.f;

    {
        const __bf16* kptr = kbase;
#if HAS_TDM
        const __bf16* vtile = vB + (size_t)b * 128 * 8192;   // linear 16KB tiles
        if (wave == 0)
            tdm_load_vtile(vtile, (unsigned)(uintptr_t)&vlds[0][0]);
#else
        const __bf16* vptr = vB + (((size_t)b * 128) * 16 * 32 + lane) * 16;
#endif
        for (int jb = 0; jb < 128; ++jb) {
#if HAS_TDM
            const int buf = jb & 1;
            if (wave == 0)
                __builtin_amdgcn_s_wait_tensorcnt(0);        // tile[jb] landed
            __syncthreads();                                 // visible to all waves
            if (wave == 0 && jb < 127)                       // overlap next DMA
                tdm_load_vtile(vtile + (size_t)(jb + 1) * 8192,
                               (unsigned)(uintptr_t)&vlds[buf ^ 1][0]);
            const __bf16* vsrc = &vlds[buf][lane * 16];
#endif
            const v16bf kb0 = *(const v16bf*)kptr;
            const v16bf kb1 = *(const v16bf*)(kptr + 512);
            __builtin_prefetch(kptr + 1024, 0, 0);           // next K tiles
            kptr += 1024;

            v8f e0 = __builtin_amdgcn_wmma_f32_16x16x32_bf16(false, qa, false, kb0,
                                                             (short)0, vzero, false, false);
            v8f e1 = __builtin_amdgcn_wmma_f32_16x16x32_bf16(false, qa, false, kb1,
                                                             (short)0, vzero, false, false);
#pragma unroll
            for (int r = 0; r < 8; ++r) {
                const float p0 = __expf(e0[r] - mloc[r]);
                const float p1 = __expf(e1[r] - mloc[r]);
                lsum[r] += p0 + p1;
                *a0[r] = (__bf16)p0;                         // ds_store_b16 scatter
                *a1[r] = (__bf16)p1;
            }
            asm volatile("s_wait_dscnt 0" ::: "memory");
            const v16bf pa = *(const v16bf*)(myp + lane * 16);

#pragma unroll
            for (int ct = 0; ct < 16; ++ct) {
#if HAS_TDM
                const v16bf vb = *(const v16bf*)(vsrc + (size_t)ct * 512);
#else
                const v16bf vb = *(const v16bf*)(vptr + (size_t)ct * 512);
#endif
                acc[ct] = __builtin_amdgcn_wmma_f32_16x16x32_bf16(false, pa, false, vb,
                                                                  (short)0, acc[ct],
                                                                  false, false);
            }
#if !HAS_TDM
            vptr += 16 * 512;
#endif
        }
    }

    // batched cross-lane reduction of the softmax denominators
#pragma unroll
    for (int s = 1; s < 16; s <<= 1) {
        float t[8];
#pragma unroll
        for (int r = 0; r < 8; ++r) t[r] = __shfl_xor(lsum[r], s, 32);
#pragma unroll
        for (int r = 0; r < 8; ++r) lsum[r] += t[r];
    }

    // epilogue: normalize, gamma * out + x
    const float g = gamma[0];
    float inv[8];
#pragma unroll
    for (int r = 0; r < 8; ++r) inv[r] = 1.f / lsum[r];

    const int i0 = it * 16 + 8 * group;
#pragma unroll
    for (int ct = 0; ct < 16; ++ct) {
        const int c = ct * 16 + lm;
        const size_t basecol = ((size_t)b * CCH + c) * (size_t)NTOK;
#pragma unroll
        for (int r = 0; r < 8; ++r) {
            const size_t idx = basecol + (size_t)(i0 + r);
            out[idx] = g * (acc[ct][r] * inv[r]) + x[idx];
        }
    }
}

// ---------------------------------------------------------------------------
extern "C" void kernel_launch(void* const* d_in, const int* in_sizes, int n_in,
                              void* d_out, int out_size, void* d_ws, size_t ws_size,
                              hipStream_t stream)
{
    const float* x     = (const float*)d_in[0];
    const float* wq    = (const float*)d_in[1];
    const float* bq    = (const float*)d_in[2];
    const float* wk    = (const float*)d_in[3];
    const float* bk    = (const float*)d_in[4];
    const float* wv    = (const float*)d_in[5];
    const float* bv    = (const float*)d_in[6];
    const float* gamma = (const float*)d_in[7];
    float* out = (float*)d_out;

    // workspace layout (bf16):
    //   qA : 4*256*32*16        = 524288  elems (1 MB)
    //   kB : 4*256*32*16        = 524288  elems (1 MB)
    //   vB : 4*128*16*32*16     = 4194304 elems (8 MB)
    __bf16* qA = (__bf16*)d_ws;
    __bf16* kB = qA + (size_t)NB * 256 * 32 * 16;
    __bf16* vB = kB + (size_t)NB * 256 * 32 * 16;

    // 1) projections + operand swizzle: one block per (batch, output row)
    proj_swizzle_kernel<<<NB * 320, 256, 0, stream>>>(x, wq, bq, wk, bk, wv, bv,
                                                      qA, kB, vB);

    // 2) fused two-pass attention + residual: 4 waves/block, 64 rows/block,
    //    V tiles DMA'd by the Tensor Data Mover, double-buffered in LDS
    flash_attn_kernel<<<NB * 64, 128, 0, stream>>>(qA, kB, vB, x, gamma, out);
}
